// ContrLoss_69071664054499
// MI455X (gfx1250) — matrix operand order repaired and checked by reading, not compiled, not run on hardware
//
#include <hip/hip_runtime.h>
#include <stdint.h>

// ---------------------------------------------------------------------------
// Problem constants (fixed by the reference: B=4, F=256, H=W=64, C=19, S=256)
// ---------------------------------------------------------------------------
#define NPIX   16384                 // B*H*W
#define FDIM   256                   // feature dim (K of the GEMM)
#define NCLS   19
#define SHALF  256                   // S
#define MCOLS  (NCLS * 2 * SHALF)    // 9728 memory entries (N of the GEMM)
#define NCHUNK (MCOLS / 16)          // 608 column chunks of 16
#define TEMP_INV 2.0f                // 1/temp, temp = 0.5
#define EPSV   1e-12f

// workspace layout (bytes)
#define WS_FEAT 0                                      // f16 [NPIX][FDIM]  : 8,388,608 B
#define WS_MEM  ((size_t)NPIX * FDIM * 2)              // f16 [MCOLS][FDIM] : 4,980,736 B
#define WS_ACC  (WS_MEM + (size_t)MCOLS * FDIM * 2)    // float contrib[32] + int cnt[32]

typedef __attribute__((ext_vector_type(16))) _Float16 v16h;
typedef __attribute__((ext_vector_type(8)))  _Float16 v8h;
typedef __attribute__((ext_vector_type(8)))  float    v8f;

// ---------------------------------------------------------------------------
// CDNA5 async-copy helpers (ASYNCcnt-tracked global->LDS DMA, no VGPR return)
// INST_OFFSET is added to BOTH the LDS and the global address (ISA 10.7), so
// one VGPR address pair serves all 16 rows of a panel via immediate offsets.
// ---------------------------------------------------------------------------
__device__ __forceinline__ unsigned lds_addr(const void* p) {
    // generic LDS pointers carry the LDS byte offset in the low 32 bits
    return (unsigned)(uintptr_t)p;
}
#define ASYNC_ROW(l, g, imm)                                                   \
    asm volatile("global_load_async_to_lds_b128 %0, %1, off offset:" #imm      \
                 :: "v"(l), "v"(g) : "memory")
__device__ __forceinline__ void async_panel16(unsigned l, uint64_t g) {
    ASYNC_ROW(l, g, 0);    ASYNC_ROW(l, g, 512);  ASYNC_ROW(l, g, 1024);
    ASYNC_ROW(l, g, 1536); ASYNC_ROW(l, g, 2048); ASYNC_ROW(l, g, 2560);
    ASYNC_ROW(l, g, 3072); ASYNC_ROW(l, g, 3584); ASYNC_ROW(l, g, 4096);
    ASYNC_ROW(l, g, 4608); ASYNC_ROW(l, g, 5120); ASYNC_ROW(l, g, 5632);
    ASYNC_ROW(l, g, 6144); ASYNC_ROW(l, g, 6656); ASYNC_ROW(l, g, 7168);
    ASYNC_ROW(l, g, 7680);
}
__device__ __forceinline__ void wait_asynccnt_le16() {
    asm volatile("s_wait_asynccnt 0x10" ::: "memory");
}
__device__ __forceinline__ void wait_asynccnt_0() {
    asm volatile("s_wait_asynccnt 0x0" ::: "memory");
}

// ---------------------------------------------------------------------------
// K0: zero the per-class accumulators
// ---------------------------------------------------------------------------
__global__ void k_init(float* __restrict__ contrib, int* __restrict__ cnt) {
    int t = threadIdx.x;
    if (t < 32) { contrib[t] = 0.0f; cnt[t] = 0; }
}

// ---------------------------------------------------------------------------
// K1: unit-normalize memory-bank rows (fp32 math) -> f16 row-major [MCOLS][F]
// ---------------------------------------------------------------------------
__global__ void __launch_bounds__(32)
k_norm_mem(const float* __restrict__ mem, _Float16* __restrict__ memN) {
    const int row  = blockIdx.x;
    const int lane = threadIdx.x;
    const float* src = mem + (size_t)row * FDIM;
    float v[8];
    float ss = 0.0f;
#pragma unroll
    for (int k = 0; k < 8; ++k) { v[k] = src[lane + 32 * k]; ss += v[k] * v[k]; }
#pragma unroll
    for (int off = 16; off >= 1; off >>= 1) ss += __shfl_xor(ss, off, 32);
    const float rn = 1.0f / sqrtf(ss);
    _Float16* dst = memN + (size_t)row * FDIM;
#pragma unroll
    for (int k = 0; k < 8; ++k) dst[lane + 32 * k] = (_Float16)(v[k] * rn);
}

// ---------------------------------------------------------------------------
// K2: transpose (B,F,H,W) -> (N,F), unit-normalize, store f16 row-major
// ---------------------------------------------------------------------------
__global__ void __launch_bounds__(32)
k_norm_feat(const float* __restrict__ pred, _Float16* __restrict__ featN) {
    const int n    = blockIdx.x;          // pixel index = b*4096 + h*64 + w
    const int lane = threadIdx.x;
    const int b    = n >> 12;
    const int hw   = n & 4095;
    const float* src = pred + (size_t)b * FDIM * 4096 + hw;
    float v[8];
    float ss = 0.0f;
#pragma unroll
    for (int k = 0; k < 8; ++k) {
        v[k] = src[(size_t)(lane + 32 * k) * 4096];
        ss += v[k] * v[k];
    }
#pragma unroll
    for (int off = 16; off >= 1; off >>= 1) ss += __shfl_xor(ss, off, 32);
    const float rn = 1.0f / sqrtf(ss);
    _Float16* dst = featN + (size_t)n * FDIM;
#pragma unroll
    for (int k = 0; k < 8; ++k) dst[lane + 32 * k] = (_Float16)(v[k] * rn);
}

// ---------------------------------------------------------------------------
// K3: fused WMMA contrastive kernel with async-LDS double buffering.
//
//   One wave per 16-pixel tile. The B panel for column chunk ch+1
//   (16 rows x 512 B = 8 KB) is prefetched by the async engine into LDS
//   while 8 x v_wmma_f32_16x16x32_f16 consume the current panel from LDS.
//   All 8 B fragments are preloaded into registers so the 16 ds_load_b128
//   issue as a clause and overlap the WMMA chain.
//
//   WMMA layouts used by this kernel (wave32, lr = lane&15, hi = lane>>4):
//     A 16x32 f16 : row M = lr, K = 16*hi + 0..15   (32 contiguous bytes)
//     B 32x16 f16 : col N = lr, K = 16*hi + 0..15   (32 contiguous bytes)
//     C 16x16 f32 : col N = lr, row M = i + 8*hi for VGPR i
// ---------------------------------------------------------------------------
__global__ void __launch_bounds__(32)
k_main(const _Float16* __restrict__ featN, const _Float16* __restrict__ memN,
       const int* __restrict__ labels, const uint8_t* __restrict__ mask,
       const int* __restrict__ wm,
       float* __restrict__ contrib, int* __restrict__ cnt) {
    __shared__ __align__(32) _Float16 panel[2][16][FDIM];  // 2 x 8 KB B panels
    __shared__ float pos[16][SHALF];                       // 16 KB pos staging
    __shared__ float trash[32];                            // sink for OOB stores
    __shared__ float tot_l[16];
    __shared__ float blk_l[16];

    const int lane    = threadIdx.x;
    const int lr      = lane & 15;
    const int hi      = lane >> 4;
    const int rowbase = blockIdx.x * 16;

    // per-lane metadata for the 8 accumulator rows (M = i + 8*hi)
    int cstart[8], hstart[8];
#pragma unroll
    for (int i = 0; i < 8; ++i) {
        const int g = rowbase + i + 8 * hi;
        const int c = labels[g];
        cstart[i] = c * (2 * SHALF);
        hstart[i] = cstart[i] + (wm[g] == 1 ? 0 : SHALF);   // net=0 <-> wm==1 <-> first half
    }

    // hoist the A tile (16 pixels x 256 K) into registers: 8 x v16h
    v16h a[8];
    {
        const _Float16* ap = featN + (size_t)(rowbase + lr) * FDIM;
#pragma unroll
        for (int kb = 0; kb < 8; ++kb) {
            const v8h lo  = *(const v8h*)(ap + kb * 32 + 16 * hi);
            const v8h hi8 = *(const v8h*)(ap + kb * 32 + 16 * hi + 8);
            v16h av;
#pragma unroll
            for (int j = 0; j < 8; ++j) { av[j] = lo[j]; av[j + 8] = hi8[j]; }
            a[kb] = av;
        }
    }

    const unsigned laneoff = (unsigned)lane * 16u;
    const unsigned pb[2]   = { lds_addr(&panel[0][0][0]) + laneoff,
                               lds_addr(&panel[1][0][0]) + laneoff };
    uint64_t gnext = (uint64_t)(uintptr_t)memN + laneoff;   // next panel, +lane*16

    // prefetch panel for chunk 0
    async_panel16(pb[0], gnext);
    gnext += 16u * 512u;

    float tot[8] = {};
    float blk[8] = {};

    for (int ch = 0; ch < NCHUNK; ++ch) {
        const int mbase = ch * 16;

        if (ch + 1 < NCHUNK) {
            async_panel16(pb[(ch & 1) ^ 1], gnext);   // prefetch next panel
            gnext += 16u * 512u;
            wait_asynccnt_le16();   // current panel resident; next in flight
        } else {
            wait_asynccnt_0();      // last panel must be resident
        }

        // preload all 8 B fragments (16 ds_load_b128, clause + staged waits)
        const _Float16* bp = &panel[(ch & 1)][lr][16 * hi];
        v16h bf[8];
#pragma unroll
        for (int kb = 0; kb < 8; ++kb) bf[kb] = *(const v16h*)(bp + kb * 32);

        v8f acc = {};
#pragma unroll
        for (int kb = 0; kb < 8; ++kb) {
            acc = __builtin_amdgcn_wmma_f32_16x16x32_f16(
                /*neg_a=*/false, a[kb], /*neg_b=*/false, bf[kb],
                /*c_mod=*/(short)0, acc, /*reuse_a=*/false, /*reuse_b=*/false);
        }

#pragma unroll
        for (int i = 0; i < 8; ++i) {
            const float e = __expf(acc[i] * TEMP_INV);   // E = exp(cos/temp)
            tot[i] += e;                                 // row total
            const int rel = mbase - cstart[i];
            if ((unsigned)rel < (unsigned)(2 * SHALF)) blk[i] += e;   // block_sum
            // branchless pos stash: select real slot or trash sink
            const int rh = mbase - hstart[i];
            float* dst = ((unsigned)rh < (unsigned)SHALF)
                             ? &pos[i + 8 * hi][rh + lr] : &trash[lane];
            *dst = e;
        }
    }

    // reduce tot/blk across the 16 columns held by lanes
    if (lane < 16) { tot_l[lane] = 0.0f; blk_l[lane] = 0.0f; }
    __syncthreads();
#pragma unroll
    for (int i = 0; i < 8; ++i) {
        atomicAdd(&tot_l[i + 8 * hi], tot[i]);
        atomicAdd(&blk_l[i + 8 * hi], blk[i]);
    }
    __syncthreads();

    // per masked pixel: term = sum_s -log(pos/(pos+down+eps)+eps)
    for (int p = 0; p < 16; ++p) {
        const int g = rowbase + p;
        if (!mask[g]) continue;
        const float down = tot_l[p] - blk_l[p];
        float t = 0.0f;
#pragma unroll
        for (int k = 0; k < 8; ++k) {
            const float pv = pos[p][lane + 32 * k];
            t -= __logf(pv / (pv + down + EPSV) + EPSV);
        }
#pragma unroll
        for (int off = 16; off >= 1; off >>= 1) t += __shfl_xor(t, off, 32);
        if (lane == 0) {
            const int c = labels[g];
            atomicAdd(&contrib[c], t);
            atomicAdd(&cnt[c], 1);
        }
    }
}

// ---------------------------------------------------------------------------
// K4: final scalar reduction over the 19 classes
// ---------------------------------------------------------------------------
__global__ void k_final(const float* __restrict__ contrib,
                        const int* __restrict__ cnt, float* __restrict__ out) {
    if (threadIdx.x == 0 && blockIdx.x == 0) {
        float loss = 0.0f, cc = 0.0f;
        for (int c = 0; c < NCLS; ++c) {
            const int n = cnt[c];
            if (n > 0) {
                loss += contrib[c] / ((float)n * (float)SHALF);
                cc += 1.0f;
            }
        }
        out[0] = loss / fmaxf(cc, 1.0f);
    }
}

// ---------------------------------------------------------------------------
// launch
// ---------------------------------------------------------------------------
extern "C" void kernel_launch(void* const* d_in, const int* in_sizes, int n_in,
                              void* d_out, int out_size, void* d_ws, size_t ws_size,
                              hipStream_t stream) {
    (void)in_sizes; (void)n_in; (void)out_size; (void)ws_size;

    const float*   memory_bank = (const float*)d_in[0];   // (19, 512, 256) f32
    const float*   pred_rep    = (const float*)d_in[1];   // (4, 256, 64, 64) f32
    const int*     labels      = (const int*)d_in[2];     // (4, 64, 64) i32
    const uint8_t* mask        = (const uint8_t*)d_in[3]; // (4, 64, 64) bool
    const int*     wm          = (const int*)d_in[4];     // (4, 64, 64) i32
    // d_in[5] = num_classes (scalar, == 19, compile-time constant here)

    char* ws = (char*)d_ws;
    _Float16* featN   = (_Float16*)(ws + WS_FEAT);
    _Float16* memN    = (_Float16*)(ws + WS_MEM);
    float*    contrib = (float*)(ws + WS_ACC);
    int*      cnt     = (int*)(ws + WS_ACC + 32 * sizeof(float));

    k_init     <<<1,          32, 0, stream>>>(contrib, cnt);
    k_norm_mem <<<MCOLS,      32, 0, stream>>>(memory_bank, memN);
    k_norm_feat<<<NPIX,       32, 0, stream>>>(pred_rep, featN);
    k_main     <<<NPIX / 16,  32, 0, stream>>>(featN, memN, labels, mask, wm,
                                               contrib, cnt);
    k_final    <<<1,           1, 0, stream>>>(contrib, cnt, (float*)d_out);
}
